// LogicConv2d_4440996184572
// MI455X (gfx1250) — matrix operand order repaired and checked by reading, not compiled, not run on hardware
//
#include <hip/hip_runtime.h>

typedef __attribute__((ext_vector_type(16))) _Float16 v16h;
typedef __attribute__((ext_vector_type(8)))  float    v8f;

#define B_    16
#define C_    3
#define H_    64
#define W_    64
#define K_    32
#define S_    16
#define RF_   5
#define OUTW_ 60
#define P_    3600      // 60*60
#define HW_   4096      // H*W
#define CHW_  12288     // C*H*W
#define NGATES 992      // K_ * (16+8+4+2+1)
#define NWAVES 62       // NGATES / 16
#define BG_   4         // batches staged per LDS round
#define RMAX_ 10        // max input rows a 256-pixel block can touch

// The 16 soft-gate mixing rows (logit basis -> [c0,c1,c2,c3])
__device__ __constant__ float M_TBL[16][4] = {
    {0,0,0,0},{0,0,0,1},{0,1,0,-1},{0,1,0,0},
    {0,0,1,-1},{0,0,1,0},{0,1,1,-2},{0,1,1,-1},
    {1,-1,-1,1},{1,-1,-1,2},{1,0,-1,0},{1,0,-1,1},
    {1,-1,0,0},{1,-1,0,1},{1,0,0,-1},{1,0,0,0}};

// -------------------------------------------------------------------------
// Kernel 1: coef[g][0..3] = softmax(w_level[g]) @ _M  via v_wmma_f32_16x16x32_f16
// One wave per 16 gates. A = probs (16x16 padded to 16x32), B = _M (16x4 padded
// to 32x16). Register packing follows the CDNA5 WMMA VGPR layout tables.
// -------------------------------------------------------------------------
__global__ __launch_bounds__(32) void coef_kernel(
    const float* __restrict__ w0, const float* __restrict__ w1,
    const float* __restrict__ w2, const float* __restrict__ w3,
    const float* __restrict__ w4, float* __restrict__ coef)
{
    const int wid = blockIdx.x;
    const float* w; int gInLvl, gGlobalBase;
    if (wid < 32)      { w = w0; gInLvl = (wid     ) * 16; gGlobalBase =   0 + gInLvl; }
    else if (wid < 48) { w = w1; gInLvl = (wid - 32) * 16; gGlobalBase = 512 + gInLvl; }
    else if (wid < 56) { w = w2; gInLvl = (wid - 48) * 16; gGlobalBase = 768 + gInLvl; }
    else if (wid < 60) { w = w3; gInLvl = (wid - 56) * 16; gGlobalBase = 896 + gInLvl; }
    else               { w = w4; gInLvl = (wid - 60) * 16; gGlobalBase = 960 + gInLvl; }

    const int lane = threadIdx.x;   // 0..31
    const int row  = lane & 15;     // A-matrix row M (gate within the 16-gate tile)
    const int half = lane >> 4;     // 0 -> logits 0..7 ; 1 -> logits 8..15

    // ---- softmax over this gate's 16 logits (split across lane L and L+16) ----
    const float* wl = w + (size_t)(gInLvl + row) * 16 + half * 8;
    float l0[8];
#pragma unroll
    for (int e = 0; e < 8; ++e) l0[e] = wl[e];

    float m = l0[0];
#pragma unroll
    for (int e = 1; e < 8; ++e) m = fmaxf(m, l0[e]);
    m = fmaxf(m, __shfl_xor(m, 16, 32));

    float s = 0.f, pr[8];
#pragma unroll
    for (int e = 0; e < 8; ++e) { pr[e] = __expf(l0[e] - m); s += pr[e]; }
    s += __shfl_xor(s, 16, 32);
    const float inv = 1.0f / s;

    // ---- A: 16x32 f16 (K=0..15 live, K=16..31 zero padding) ----
    v16h A;
#pragma unroll
    for (int e = 0; e < 8; ++e)  A[e] = (_Float16)(pr[e] * inv);
#pragma unroll
    for (int e = 8; e < 16; ++e) A[e] = (_Float16)0.f;

    // ---- B: 32x16 f16, B[k][n] = _M[k][n] for k<16,n<4 else 0 ----
    const int col = lane & 15;
    const float bm = (col < 4) ? 1.f : 0.f;
    const int cc = col & 3;
    v16h Bv;
#pragma unroll
    for (int e = 0; e < 16; ++e) Bv[e] = (_Float16)(e < 8 ? M_TBL[half * 8 + e][cc] * bm : 0.f);

    v8f acc = {};
    v8f d = __builtin_amdgcn_wmma_f32_16x16x32_f16(
        /*neg_a=*/false, A, /*neg_b=*/false, Bv,
        /*c_mod=*/(short)0, acc, /*reuse_a=*/false, /*reuse_b=*/false);

    // D layout: lane L -> column N=L&15 ; VGPR r -> row M = (L>>4)*8 + r.
    if (col < 4) {
#pragma unroll
        for (int r = 0; r < 8; ++r) {
            int gm = half * 8 + r;
            coef[(size_t)(gGlobalBase + gm) * 4 + col] = d[r];
        }
    }
}

// -------------------------------------------------------------------------
// Kernel 2: LDS-staged gather + 5-level soft-gate tree.
// One block = one k x 256 consecutive output pixels. The block's gather
// footprint in x is <= 3ch x 10rows x 64cols = 7.7KB per batch; stage 4
// batches at a time in LDS (30KB) with coalesced float4 loads, then do all
// 32 gathers/batch from LDS. Indices are batch-invariant: loaded once as
// int4 and converted to slab-relative LDS cell offsets.
// -------------------------------------------------------------------------
__global__ __launch_bounds__(256) void logic_tree_kernel(
    const float* __restrict__ x,
    const int* __restrict__ ah, const int* __restrict__ aw, const int* __restrict__ ac,
    const int* __restrict__ bh, const int* __restrict__ bw, const int* __restrict__ bc,
    const float* __restrict__ coef, float* __restrict__ out)
{
    __shared__ float sc[31 * 4];                                    // gate coefs for this k
    __shared__ __align__(16) float slab[BG_ * C_ * RMAX_ * W_];     // 30 KB x-slab

    const int k = blockIdx.y;
    const int t = threadIdx.x;
    const int p = blockIdx.x * 256 + t;
    const bool active = (p < P_);
    const int pc = active ? p : (P_ - 1);     // clamp: keep all threads in barriers

    // Block's input-row window (depends only on blockIdx.x).
    const int p0   = blockIdx.x * 256;
    const int pend = min(p0 + 255, P_ - 1);
    const int r0   = p0 / OUTW_;
    const int rows = pend / OUTW_ + (RF_ - 1) - r0 + 1;   // 5..10
    const int bstr = C_ * rows * W_;                      // floats per batch slot in slab

    // Stage this k's coefs into LDS. Node enum: l0:0-15, l1:16-23, l2:24-27,
    // l3:28-29, l4:30. Gate id in coef[] = lvlBase[l] + n*K + k.
    if (t < 124) {
        int node = t >> 2, i = t & 3;
        int l, n;
        if      (node < 16) { l = 0; n = node;      }
        else if (node < 24) { l = 1; n = node - 16; }
        else if (node < 28) { l = 2; n = node - 24; }
        else if (node < 30) { l = 3; n = node - 28; }
        else                { l = 4; n = 0;         }
        const int lvlBase[5] = {0, 512, 768, 896, 960};
        sc[t] = coef[(size_t)(lvlBase[l] + n * K_ + k) * 4 + i];
    }

    // Coalesced 128-bit index loads (batch-invariant) -> slab-relative cells.
    const size_t ib = ((size_t)k * P_ + pc) * S_;
    int cellA[S_], cellB[S_];
#pragma unroll
    for (int q = 0; q < 4; ++q) {
        int4 vh = ((const int4*)(ah + ib))[q];
        int4 vw = ((const int4*)(aw + ib))[q];
        int4 vc = ((const int4*)(ac + ib))[q];
        cellA[4*q+0] = (vc.x * rows + vh.x - r0) * W_ + vw.x;
        cellA[4*q+1] = (vc.y * rows + vh.y - r0) * W_ + vw.y;
        cellA[4*q+2] = (vc.z * rows + vh.z - r0) * W_ + vw.z;
        cellA[4*q+3] = (vc.w * rows + vh.w - r0) * W_ + vw.w;
    }
#pragma unroll
    for (int q = 0; q < 4; ++q) {
        int4 vh = ((const int4*)(bh + ib))[q];
        int4 vw = ((const int4*)(bw + ib))[q];
        int4 vc = ((const int4*)(bc + ib))[q];
        cellB[4*q+0] = (vc.x * rows + vh.x - r0) * W_ + vw.x;
        cellB[4*q+1] = (vc.y * rows + vh.y - r0) * W_ + vw.y;
        cellB[4*q+2] = (vc.z * rows + vh.z - r0) * W_ + vw.z;
        cellB[4*q+3] = (vc.w * rows + vh.w - r0) * W_ + vw.w;
    }

    const int lines = BG_ * C_ * rows;      // 64-float lines per batch group
    const int nf4   = lines * 16;           // float4 units to stage

    for (int bg = 0; bg < B_; bg += BG_) {
        __syncthreads();   // previous group's LDS reads done (also covers sc on bg==0)

        // Stage batch group [bg, bg+BG_) : fully coalesced b128 loads -> LDS.
        for (int e = t; e < nf4; e += 256) {
            int seg  = e & 15;
            int line = e >> 4;
            int r  = line % rows;
            int cb = line / rows;           // c + C_*b
            int c  = cb % C_;
            int b  = cb / C_;
            const float4 v = ((const float4*)(x +
                (size_t)(bg + b) * CHW_ + c * HW_ + (size_t)(r0 + r) * W_))[seg];
            ((float4*)slab)[line * 16 + seg] = v;
        }
        __syncthreads();

        // Prefetch next group's lines into cache while we compute (CDNA5
        // global_prefetch_b8; lines <= 120 so at most one per thread).
        if (bg + BG_ < B_ && t < lines) {
            int r  = t % rows;
            int cb = t / rows;
            int c  = cb % C_;
            int b  = cb / C_;
            __builtin_prefetch(x + (size_t)(bg + BG_ + b) * CHW_ + c * HW_ +
                               (size_t)(r0 + r) * W_, 0, 2);
        }

        // Evaluate the gate tree for each batch in the group (LDS gathers).
        for (int bl = 0; bl < BG_; ++bl) {
            const float* xs = slab + bl * bstr;
            float val[S_];
#pragma unroll
            for (int n = 0; n < 16; ++n) {
                float a  = xs[cellA[n]];
                float bb = xs[cellB[n]];
                const float* c = &sc[n * 4];
                val[n] = fmaf(c[3], a * bb, fmaf(c[2], bb, fmaf(c[1], a, c[0])));
            }
            int nodeBase = 16;
#pragma unroll
            for (int l = 1, width = 8; l <= 4; ++l, width >>= 1) {
#pragma unroll
                for (int n = 0; n < width; ++n) {
                    float a  = val[2*n];
                    float bb = val[2*n + 1];
                    const float* c = &sc[(nodeBase + n) * 4];
                    val[n] = fmaf(c[3], a * bb, fmaf(c[2], bb, fmaf(c[1], a, c[0])));
                }
                nodeBase += width;
            }
            if (active)
                out[((size_t)(bg + bl) * K_ + k) * P_ + p] = val[0];
        }
    }
}

extern "C" void kernel_launch(void* const* d_in, const int* in_sizes, int n_in,
                              void* d_out, int out_size, void* d_ws, size_t ws_size,
                              hipStream_t stream) {
    const float* x  = (const float*)d_in[0];
    const int*   ah = (const int*)d_in[1];
    const int*   aw = (const int*)d_in[2];
    const int*   ac = (const int*)d_in[3];
    const int*   bh = (const int*)d_in[4];
    const int*   bw = (const int*)d_in[5];
    const int*   bc = (const int*)d_in[6];
    const float* w0 = (const float*)d_in[7];
    const float* w1 = (const float*)d_in[8];
    const float* w2 = (const float*)d_in[9];
    const float* w3 = (const float*)d_in[10];
    const float* w4 = (const float*)d_in[11];
    float* out  = (float*)d_out;
    float* coef = (float*)d_ws;   // NGATES*4 floats = 15.5 KB of scratch

    coef_kernel<<<NWAVES, 32, 0, stream>>>(w0, w1, w2, w3, w4, coef);

    dim3 grid((P_ + 255) / 256, K_);
    logic_tree_kernel<<<grid, 256, 0, stream>>>(x, ah, aw, ac, bh, bw, bc, coef, out);
}
